// FalconMambaMixer_63239098466599
// MI455X (gfx1250) — matrix operand order repaired and checked by reading, not compiled, not run on hardware
//
#include <hip/hip_runtime.h>

// ---------------------------------------------------------------------------
// Problem constants (match reference)
// ---------------------------------------------------------------------------
#define Hdim   4096
#define DIN    8192
#define Lseq   1024
#define Bsz    2
#define Rdim   256
#define Nst    16
#define Kconv  4
#define Pdim   (2 * DIN)          // 16384
#define Mtok   (Bsz * Lseq)       // 2048
#define RNdim  (Rdim + 2 * Nst)   // 288
#define EPSrms 1e-6f

// ---------------------------------------------------------------------------
// WMMA vector types (CDNA5 / gfx1250, wave32)
// ---------------------------------------------------------------------------
typedef __attribute__((ext_vector_type(16))) __bf16    v16bf;
typedef __attribute__((ext_vector_type(8)))  float     v8f;
typedef __attribute__((ext_vector_type(4)))  unsigned  v4u;
typedef __attribute__((ext_vector_type(4)))  int       v4i;

#if __has_builtin(__builtin_amdgcn_global_load_async_to_lds_b128)
#define HAS_ASYNC_LDS 1
#endif

__device__ __forceinline__ unsigned pack_bf16(float x, float y) {
    unsigned ux = __builtin_bit_cast(unsigned, x) + 0x8000u;   // round-half-up
    unsigned uy = __builtin_bit_cast(unsigned, y) + 0x8000u;
    return (uy & 0xFFFF0000u) | (ux >> 16);
}
__device__ __forceinline__ unsigned short f2bf(float x) {
    unsigned u = __builtin_bit_cast(unsigned, x) + 0x8000u;
    return (unsigned short)(u >> 16);
}
__device__ __forceinline__ float bf2f(unsigned short h) {
    return __builtin_bit_cast(float, (unsigned)h << 16);
}

// two 16-byte LDS reads -> one 16 x bf16 fragment
__device__ __forceinline__ v16bf ld_frag(const unsigned* s, int off0, int off1) {
    union { v4u u[2]; v16bf f; } r;
    r.u[0] = *(const v4u*)(s + off0);
    r.u[1] = *(const v4u*)(s + off1);
    return r.f;
}

#ifdef HAS_ASYNC_LDS
// async global -> LDS 16B copy, tracked by ASYNCcnt
// (builtin params are generic int4* per clang diagnostic; src first, dst second)
__device__ __forceinline__ void async_copy_b128(unsigned* lds, const unsigned short* g) {
    __builtin_amdgcn_global_load_async_to_lds_b128((v4i*)g, (v4i*)lds, 0, 0);
}
template <int N>
__device__ __forceinline__ void wait_asynccnt() {
#if __has_builtin(__builtin_amdgcn_s_wait_asynccnt)
    __builtin_amdgcn_s_wait_asynccnt(N);
#else
    asm volatile("s_wait_asynccnt %0" :: "n"(N));
#endif
}
#endif

// ---------------------------------------------------------------------------
// f32 -> packed bf16 pair conversion (one-time, before the GEMMs)
// ---------------------------------------------------------------------------
__global__ __launch_bounds__(256)
void cvt_f32_bf16_kernel(const float* __restrict__ src, unsigned* __restrict__ dst,
                         size_t npairs)
{
    size_t i = (size_t)blockIdx.x * 256 + threadIdx.x;
    if (i >= npairs) return;
    float2 v = ((const float2*)src)[i];
    dst[i] = pack_bf16(v.x, v.y);
}

// ---------------------------------------------------------------------------
// NT GEMM, all-bf16 operands:  C[MxN](f32) = A[MxK] * B[NxK]^T
//   M % 128 == 0, K % 64 == 0, N guarded.
// Block: 256 thr = 8 waves in 4(M) x 2(N); wave tile 32x64 -> 2x4 accumulators.
// Async path: LDS double-buffered 128x32 tiles filled by
//   GLOBAL_LOAD_ASYNC_TO_LDS_B128, pipelined one tile ahead (ASYNCcnt).
// Fallback : register double-buffered 128x64 tiles (round-2 proven path).
// ---------------------------------------------------------------------------
__global__ __launch_bounds__(256)
void gemm_nt_bf16_wmma(const unsigned short* __restrict__ A,
                       const unsigned short* __restrict__ B,
                       float* __restrict__ C, int M, int N, int K)
{
    const int tid  = threadIdx.x;
    const int lane = tid & 31;
    const int wave = tid >> 5;      // 0..7
    const int wrow = wave >> 1;     // 0..3 : M offset 32*wrow
    const int wcol = wave & 1;      // 0..1 : N offset 64*wcol
    const int half = lane >> 4;
    const int l16  = lane & 15;

    const int m0 = blockIdx.y * 128;
    const int n0 = blockIdx.x * 128;

    const v8f vz = {0.f, 0.f, 0.f, 0.f, 0.f, 0.f, 0.f, 0.f};
    v8f acc[2][4];
#pragma unroll
    for (int i = 0; i < 2; ++i)
#pragma unroll
        for (int j = 0; j < 4; ++j) acc[i][j] = vz;

#ifdef HAS_ASYNC_LDS
    // ---------------- async pipeline: K-step 32, LDS double buffer ----------
    // row stride 20 dwords (80B): r*20 mod 64 distinct for r=0..15 -> no conflicts
    __shared__ __align__(16) unsigned As[2][128 * 20];
    __shared__ __align__(16) unsigned Bs[2][128 * 20];

    auto issue_tile = [&](int kt, int buf) {
#pragma unroll
        for (int i = 0; i < 2; ++i) {
            int p = tid + i * 256;          // 512 16B chunks per 128x32 tile
            int r = p >> 2, q = p & 3;
            async_copy_b128(&As[buf][r * 20 + q * 4],
                            A + (size_t)(m0 + r) * K + kt + q * 8);
            int j = n0 + r; if (j >= N) j = N - 1;   // clamp: acc never stored
            async_copy_b128(&Bs[buf][r * 20 + q * 4],
                            B + (size_t)j * K + kt + q * 8);
        }
    };

    issue_tile(0, 0);
    const int T = K / 32;
    for (int it = 0; it < T; ++it) {
        const int cur = it & 1;
        __syncthreads();                    // buf[cur^1] free for refill
        if (it + 1 < T) {
            issue_tile((it + 1) * 32, cur ^ 1);
            wait_asynccnt<4>();             // tile `it` landed; next still in flight
            if (it + 2 < T) {
                __builtin_prefetch(A + (size_t)(m0 + (tid >> 1)) * K + (it + 2) * 32, 0, 1);
                int pj = n0 + (tid >> 1); if (pj >= N) pj = N - 1;
                __builtin_prefetch(B + (size_t)pj * K + (it + 2) * 32, 0, 1);
            }
        } else {
            wait_asynccnt<0>();
        }
        __syncthreads();                    // all waves see tile `it`

        v16bf af[2], bfr[4];
#pragma unroll
        for (int i = 0; i < 2; ++i) {
            int base = (wrow * 32 + i * 16 + l16) * 20;
            af[i] = ld_frag(As[cur], base + half * 4, base + 8 + half * 4);
        }
#pragma unroll
        for (int j = 0; j < 4; ++j) {
            int base = (wcol * 64 + j * 16 + l16) * 20;
            bfr[j] = ld_frag(Bs[cur], base + half * 4, base + 8 + half * 4);
        }
#pragma unroll
        for (int i = 0; i < 2; ++i)
#pragma unroll
            for (int j = 0; j < 4; ++j)
                acc[i][j] = __builtin_amdgcn_wmma_f32_16x16x32_bf16(
                    false, af[i], false, bfr[j], (short)0, acc[i][j], false, false);
    }
#else
    // ---------------- fallback: register-staged K-step 64 (round-2 path) ----
    __shared__ __align__(16) unsigned As[128 * 36];
    __shared__ __align__(16) unsigned Bs[128 * 36];

    const v4u z4 = {0u, 0u, 0u, 0u};
    v4u ra[4], rb[4];
#pragma unroll
    for (int i = 0; i < 4; ++i) {
        int p = tid + i * 256;
        int r = p >> 3, q = p & 7;
        ra[i] = *(const v4u*)(A + (size_t)(m0 + r) * K + q * 8);
        int j = n0 + r;
        rb[i] = (j < N) ? *(const v4u*)(B + (size_t)j * K + q * 8) : z4;
    }

    for (int k0 = 0; k0 < K; k0 += 64) {
#pragma unroll
        for (int i = 0; i < 4; ++i) {
            int p = tid + i * 256;
            int r = p >> 3, q = p & 7;
            *(v4u*)&As[r * 36 + q * 4] = ra[i];
            *(v4u*)&Bs[r * 36 + q * 4] = rb[i];
        }
        __syncthreads();

        if (k0 + 64 < K) {
            int kn = k0 + 64;
#pragma unroll
            for (int i = 0; i < 4; ++i) {
                int p = tid + i * 256;
                int r = p >> 3, q = p & 7;
                ra[i] = *(const v4u*)(A + (size_t)(m0 + r) * K + kn + q * 8);
                int j = n0 + r;
                rb[i] = (j < N) ? *(const v4u*)(B + (size_t)j * K + kn + q * 8) : z4;
            }
            if (kn + 64 < K) {
                __builtin_prefetch(A + (size_t)(m0 + (tid >> 1)) * K + kn + 64, 0, 1);
                int pj = n0 + (tid >> 1);
                if (pj < N)
                    __builtin_prefetch(B + (size_t)pj * K + kn + 64, 0, 1);
            }
        }

#pragma unroll
        for (int s = 0; s < 2; ++s) {
            v16bf af[2], bfr[4];
#pragma unroll
            for (int i = 0; i < 2; ++i) {
                int base = (wrow * 32 + i * 16 + l16) * 36 + s * 16;
                af[i] = ld_frag(As, base + half * 4, base + 8 + half * 4);
            }
#pragma unroll
            for (int j = 0; j < 4; ++j) {
                int base = (wcol * 64 + j * 16 + l16) * 36 + s * 16;
                bfr[j] = ld_frag(Bs, base + half * 4, base + 8 + half * 4);
            }
#pragma unroll
            for (int i = 0; i < 2; ++i)
#pragma unroll
                for (int j = 0; j < 4; ++j)
                    acc[i][j] = __builtin_amdgcn_wmma_f32_16x16x32_bf16(
                        false, af[i], false, bfr[j], (short)0, acc[i][j], false, false);
        }
        __syncthreads();
    }
#endif

    // store C (f32): vgpr r -> M = r + 8*half; lane l16 -> N
#pragma unroll
    for (int i = 0; i < 2; ++i)
#pragma unroll
        for (int j = 0; j < 4; ++j) {
            int nn = n0 + wcol * 64 + j * 16 + l16;
            if (nn < N) {
                int mm = m0 + wrow * 32 + i * 16 + half * 8;
#pragma unroll
                for (int r = 0; r < 8; ++r)
                    C[(size_t)(mm + r) * N + nn] = acc[i][j][r];
            }
        }
}

// ---------------------------------------------------------------------------
// Depthwise causal conv (K=4) + SiLU over the x-half of proj -> bf16
// ---------------------------------------------------------------------------
__global__ __launch_bounds__(256)
void conv_silu_kernel(const float* __restrict__ proj, const float* __restrict__ cw,
                      const float* __restrict__ cb, unsigned short* __restrict__ xcb)
{
    size_t idx = (size_t)blockIdx.x * 256 + threadIdx.x;  // over Mtok * DIN
    int d = (int)(idx % DIN);
    size_t t = idx / DIN;
    int l = (int)(t % Lseq);
    float a = cb[d];
#pragma unroll
    for (int j = 0; j < Kconv; ++j) {
        int ll = l - (Kconv - 1) + j;
        if (ll >= 0)
            a += cw[d * Kconv + j] * proj[(t - (Kconv - 1) + j) * (size_t)Pdim + d];
    }
    float s = a / (1.f + __expf(-a));
    xcb[idx] = f2bf(s);
}

// ---------------------------------------------------------------------------
// Fused RMS norms: ts (256) -> bf16, B (16) / C (16) -> f32
// ---------------------------------------------------------------------------
__global__ __launch_bounds__(256)
void rms_kernel(const float* __restrict__ ssm, unsigned short* __restrict__ tsnb,
                float* __restrict__ Bn, float* __restrict__ Cn)
{
    int t = blockIdx.x, tid = threadIdx.x;
    __shared__ float red[256];
    __shared__ float sB, sC;
    const float* row = ssm + (size_t)t * RNdim;

    float v = row[tid];
    red[tid] = v * v;
    __syncthreads();
    for (int s = 128; s > 0; s >>= 1) {
        if (tid < s) red[tid] += red[tid + s];
        __syncthreads();
    }
    float rstd = rsqrtf(red[0] / (float)Rdim + EPSrms);
    tsnb[(size_t)t * Rdim + tid] = f2bf(v * rstd);

    if (tid == 0) {
        float s = 0.f;
        for (int i = 0; i < Nst; ++i) { float w = row[Rdim + i]; s += w * w; }
        sB = rsqrtf(s / (float)Nst + EPSrms);
    }
    if (tid == 1) {
        float s = 0.f;
        for (int i = 0; i < Nst; ++i) { float w = row[Rdim + Nst + i]; s += w * w; }
        sC = rsqrtf(s / (float)Nst + EPSrms);
    }
    __syncthreads();
    if (tid < Nst)
        Bn[(size_t)t * Nst + tid] = row[Rdim + tid] * sB;
    else if (tid < 2 * Nst)
        Cn[(size_t)t * Nst + (tid - Nst)] = row[Rdim + tid] * sC;
}

// ---------------------------------------------------------------------------
// dt = softplus(dt_raw + b_dt), in place (f32)
// ---------------------------------------------------------------------------
__global__ __launch_bounds__(256)
void dt_softplus_kernel(float* __restrict__ dt, const float* __restrict__ bdt)
{
    size_t idx = (size_t)blockIdx.x * 256 + threadIdx.x;
    int d = (int)(idx % DIN);
    float x = dt[idx] + bdt[d];
    dt[idx] = (x > 20.f) ? x : log1pf(__expf(x));
}

// ---------------------------------------------------------------------------
// Selective scan: state[16] in VGPRs, one thread per (b, d).
// Fused: y = scan + u*D; y *= silu(gate).  Writes bf16 for the final GEMM.
// ---------------------------------------------------------------------------
__global__ __launch_bounds__(128)
void scan_kernel(const float* __restrict__ dt, const unsigned short* __restrict__ xcb,
                 const float* __restrict__ proj, const float* __restrict__ Bn,
                 const float* __restrict__ Cn, const float* __restrict__ Alog,
                 const float* __restrict__ Dv, unsigned short* __restrict__ ygb)
{
    const int b   = blockIdx.y;
    const int tid = threadIdx.x;
    const int d   = blockIdx.x * 128 + tid;

    __shared__ float shB[Nst], shC[Nst];

    float a[Nst], st[Nst];
#pragma unroll
    for (int n = 0; n < Nst; ++n) {
        a[n]  = -__expf(Alog[(size_t)d * Nst + n]);
        st[n] = 0.f;
    }
    const float Dd = Dv[d];

    for (int l = 0; l < Lseq; ++l) {
        size_t t = (size_t)b * Lseq + l;
        if (tid < Nst)            shB[tid]       = Bn[t * Nst + tid];
        else if (tid < 2 * Nst)   shC[tid - Nst] = Cn[t * Nst + (tid - Nst)];
        __syncthreads();

        float dtv = dt[t * DIN + d];
        float u   = bf2f(xcb[t * DIN + d]);
        float du  = dtv * u;
        float y   = 0.f;
#pragma unroll
        for (int n = 0; n < Nst; ++n) {
            st[n] = st[n] * __expf(dtv * a[n]) + du * shB[n];
            y    += st[n] * shC[n];
        }
        float g  = proj[t * (size_t)Pdim + DIN + d];
        float sg = g / (1.f + __expf(-g));
        ygb[t * DIN + d] = f2bf((y + u * Dd) * sg);
        __syncthreads();
    }
}

// ---------------------------------------------------------------------------
// Host launch
// ---------------------------------------------------------------------------
static inline size_t align256(size_t x) { return (x + 255) & ~(size_t)255; }

extern "C" void kernel_launch(void* const* d_in, const int* in_sizes, int n_in,
                              void* d_out, int out_size, void* d_ws, size_t ws_size,
                              hipStream_t stream)
{
    const float* hs    = (const float*)d_in[0]; // (2,1024,4096)
    const float* Win   = (const float*)d_in[1]; // (16384,4096)
    const float* convw = (const float*)d_in[2]; // (8192,4)
    const float* convb = (const float*)d_in[3]; // (8192,)
    const float* Wx    = (const float*)d_in[4]; // (288,8192)
    const float* Wdt   = (const float*)d_in[5]; // (8192,256)
    const float* bdt   = (const float*)d_in[6]; // (8192,)
    const float* Alog  = (const float*)d_in[7]; // (8192,16)
    const float* Dv    = (const float*)d_in[8]; // (8192,)
    const float* Wout  = (const float*)d_in[9]; // (4096,8192)
    float* out = (float*)d_out;                 // (2,1024,4096)

    char* w = (char*)d_ws;
    auto take = [&](size_t bytes) { char* p = w; w += align256(bytes); return p; };

    float*          proj  = (float*)         take((size_t)Mtok * Pdim * 4);
    float*          ssm   = (float*)         take((size_t)Mtok * RNdim * 4);
    float*          dtb   = (float*)         take((size_t)Mtok * DIN * 4);
    float*          Bn    = (float*)         take((size_t)Mtok * Nst * 4);
    float*          Cn    = (float*)         take((size_t)Mtok * Nst * 4);
    unsigned short* hsb   = (unsigned short*)take((size_t)Mtok * Hdim * 2);
    unsigned short* Winb  = (unsigned short*)take((size_t)Pdim * Hdim * 2);
    unsigned short* Wxb   = (unsigned short*)take((size_t)RNdim * DIN * 2);
    unsigned short* Wdtb  = (unsigned short*)take((size_t)DIN * Rdim * 2);
    unsigned short* Woutb = (unsigned short*)take((size_t)Hdim * DIN * 2);
    unsigned short* xcb   = (unsigned short*)take((size_t)Mtok * DIN * 2);
    unsigned short* tsnb  = (unsigned short*)take((size_t)Mtok * Rdim * 2);
    unsigned short* ygb   = (unsigned short*)take((size_t)Mtok * DIN * 2);

    dim3 blk(256);
    auto cvt = [&](const float* s, unsigned short* dst, size_t n) {
        size_t np = n / 2;
        cvt_f32_bf16_kernel<<<(unsigned)((np + 255) / 256), blk, 0, stream>>>(
            s, (unsigned*)dst, np);
    };

    // 0. one-time f32 -> bf16 conversions of all GEMM operands
    cvt(hs,   hsb,   (size_t)Mtok * Hdim);
    cvt(Win,  Winb,  (size_t)Pdim * Hdim);
    cvt(Wx,   Wxb,   (size_t)RNdim * DIN);
    cvt(Wdt,  Wdtb,  (size_t)DIN * Rdim);
    cvt(Wout, Woutb, (size_t)Hdim * DIN);

    // 1. in-projection:  proj = hs @ Win^T   (M=2048, N=16384, K=4096)
    gemm_nt_bf16_wmma<<<dim3(Pdim / 128, Mtok / 128), blk, 0, stream>>>(
        hsb, Winb, proj, Mtok, Pdim, Hdim);

    // 2. depthwise conv + silu on the x half -> bf16
    conv_silu_kernel<<<(unsigned)(((size_t)Mtok * DIN) / 256), blk, 0, stream>>>(
        proj, convw, convb, xcb);

    // 3. ssm projection:  ssm = xc @ Wx^T   (N=288)
    gemm_nt_bf16_wmma<<<dim3((RNdim + 127) / 128, Mtok / 128), blk, 0, stream>>>(
        xcb, Wxb, ssm, Mtok, RNdim, DIN);

    // 4. RMS norms of ts / B / C
    rms_kernel<<<Mtok, blk, 0, stream>>>(ssm, tsnb, Bn, Cn);

    // 5. dt gemm:  dtb = tsn @ Wdt^T   (N=8192, K=256), then softplus(+b_dt)
    gemm_nt_bf16_wmma<<<dim3(DIN / 128, Mtok / 128), blk, 0, stream>>>(
        tsnb, Wdtb, dtb, Mtok, DIN, Rdim);
    dt_softplus_kernel<<<(unsigned)(((size_t)Mtok * DIN) / 256), blk, 0, stream>>>(
        dtb, bdt);

    // 6. selective scan, fused with x*D and silu(gate) -> bf16
    scan_kernel<<<dim3(DIN / 128, Bsz), dim3(128), 0, stream>>>(
        dtb, xcb, proj, Bn, Cn, Alog, Dv, ygb);

    // 7. out-projection:  out = yg @ Wout^T   (N=4096, K=8192)
    gemm_nt_bf16_wmma<<<dim3(Hdim / 128, Mtok / 128), blk, 0, stream>>>(
        ygb, Woutb, out, Mtok, Hdim, DIN);
}
